// TransConvLayer_10333691314434
// MI455X (gfx1250) — matrix operand order repaired and checked by reading, not compiled, not run on hardware
//
#include <hip/hip_runtime.h>
#include <stdint.h>

// ---------------------------------------------------------------------------
// TransConv layer for MI455X (gfx1250, wave32, WMMA bf16 16x16x32, f32 accum)
//
// hidden = temp0*x + q @ M,   M = sum_{i=1..K} temp[i] * kv @ kq^{i-1}
//   q  = row-softmax(x Wq^T + bq)            [N,128] (stored bf16)
//   e  = exp(x Wk^T + bk - colmax)           column softmax numerator
//   kv = (1/colsum) * E^T x ;  kq = (1/colsum) * E^T q   (128x128)
//
// Memory-bound problem (~0.6 GB traffic vs ~26 GFLOP) -> bf16 operands,
// f32 accumulation, algebraic collapse of the K-step chain into one 128x128 M.
// ---------------------------------------------------------------------------

#define N_NODES 200000
#define D 128
#define KPOW 10
#define ROWS_PER_BLK 128
#define NBLK ((N_NODES + ROWS_PER_BLK - 1) / ROWS_PER_BLK)  // 1563

typedef __attribute__((ext_vector_type(16))) __bf16 v16bf;
typedef __attribute__((ext_vector_type(8)))  float  v8f;

union BFrag { v16bf v; uint4 q[2]; unsigned short h[16]; };

#define WMMA_BF16(a, b, c) \
  __builtin_amdgcn_wmma_f32_16x16x32_bf16(false, (a), false, (b), (short)0, (c), false, false)

__device__ __forceinline__ unsigned short f2bf(float f) {
  unsigned int u = __float_as_uint(f);
  u += 0x7FFFu + ((u >> 16) & 1u);           // round-to-nearest-even
  return (unsigned short)(u >> 16);
}
__device__ __forceinline__ float bf2f(unsigned short h) {
  return __uint_as_float(((unsigned int)h) << 16);
}
__device__ __forceinline__ v8f zero8() {
  v8f z = {0.f, 0.f, 0.f, 0.f, 0.f, 0.f, 0.f, 0.f};
  return z;
}

// A fragment (16x32 bf16), ISA layout:
//  lanes 0-15 : row = base+lane,    K = kb+{0..7} and kb+{16..23}
//  lanes 16-31: row = base+lane-16, K = kb+{8..15} and kb+{24..31}
__device__ __forceinline__ v16bf load_a_g(const unsigned short* __restrict__ src,
                                          int rowBase, int kb, int lane, int maxRow) {
  int l = lane & 15, hi = lane >> 4;
  int r = rowBase + l; r = r > maxRow ? maxRow : r;
  const unsigned short* p = src + (size_t)r * D + kb + (hi ? 8 : 0);
  BFrag f;
  f.q[0] = *(const uint4*)(p);
  f.q[1] = *(const uint4*)(p + 16);
  return f.v;
}
__device__ __forceinline__ v16bf load_a_l(const unsigned short* lds,
                                          int rowBase, int kb, int lane) {
  int l = lane & 15, hi = lane >> 4;
  const unsigned short* p = lds + (rowBase + l) * D + kb + (hi ? 8 : 0);
  BFrag f;
  f.q[0] = *(const uint4*)(p);
  f.q[1] = *(const uint4*)(p + 16);
  return f.v;
}
// B fragment (32x16 bf16) from LDS holding B^T row-major ([j][k], k contiguous):
//  lanes 0-15 : col j = jb+lane,    K = kb+{0..7} and kb+{16..23}
//  lanes 16-31: col j = jb+lane-16, K = kb+{8..15} and kb+{24..31}
__device__ __forceinline__ v16bf load_b_l(const unsigned short* lds,
                                          int jb, int kb, int lane) {
  int l = lane & 15, hi = lane >> 4;
  const unsigned short* p = lds + (jb + l) * D + kb + (hi ? 8 : 0);
  BFrag f;
  f.q[0] = *(const uint4*)(p);
  f.q[1] = *(const uint4*)(p + 16);
  return f.v;
}

// Stage a 128-row bf16 block transposed into LDS: dst[j][n] = src[nBase+n][j]
__device__ __forceinline__ void stage_T(const unsigned short* __restrict__ src,
                                        unsigned short* dst, int nBase, int tid) {
  for (int idx = tid; idx < D * 16; idx += 256) {
    int n = idx >> 4, db = (idx & 15) << 3;
    int r = nBase + n; r = r >= N_NODES ? (N_NODES - 1) : r;
    uint4 v = *(const uint4*)(src + (size_t)r * D + db);
    const unsigned short* s = (const unsigned short*)&v;
#pragma unroll
    for (int j = 0; j < 8; ++j) dst[(db + j) * D + n] = s[j];
  }
}

// ---------------------------------------------------------------------------
__global__ __launch_bounds__(256) void k_convert(const float* __restrict__ x,
                                                 unsigned short* __restrict__ xb) {
  size_t i = ((size_t)blockIdx.x * 256 + threadIdx.x) * 8;
  if (i >= (size_t)N_NODES * D) return;
  float4 a = *(const float4*)(x + i);
  float4 b = *(const float4*)(x + i + 4);
  unsigned int u0 = (unsigned int)f2bf(a.x) | ((unsigned int)f2bf(a.y) << 16);
  unsigned int u1 = (unsigned int)f2bf(a.z) | ((unsigned int)f2bf(a.w) << 16);
  unsigned int u2 = (unsigned int)f2bf(b.x) | ((unsigned int)f2bf(b.y) << 16);
  unsigned int u3 = (unsigned int)f2bf(b.z) | ((unsigned int)f2bf(b.w) << 16);
  *(uint4*)(xb + i) = make_uint4(u0, u1, u2, u3);
}

__global__ void k_zero(float* p, int n) {
  int i = blockIdx.x * blockDim.x + threadIdx.x;
  if (i < n) p[i] = 0.f;
}

// ---------------------------------------------------------------------------
// Pass 1: q = row-softmax(x Wq^T + bq) (bf16 out), partial column-max of Klog
__global__ __launch_bounds__(256) void k_qk(const unsigned short* __restrict__ xb,
                                            const float* __restrict__ Wq,
                                            const float* __restrict__ bq,
                                            const float* __restrict__ Wk,
                                            const float* __restrict__ bk,
                                            unsigned short* __restrict__ qout,
                                            float* __restrict__ partmax) {
  __shared__ __align__(16) unsigned short Wql[D * D];
  __shared__ __align__(16) unsigned short Wkl[D * D];
  __shared__ float red[8 * D];
  int tid = threadIdx.x, lane = tid & 31, wave = tid >> 5;
  int l = lane & 15, hi = lane >> 4;
  for (int i = tid; i < D * D; i += 256) { Wql[i] = f2bf(Wq[i]); Wkl[i] = f2bf(Wk[i]); }
  __syncthreads();

  int rowBase = blockIdx.x * ROWS_PER_BLK + wave * 16;
  int maxRow = N_NODES - 1;
  v8f acc[8];
  float rstat[8];

  // ---- Q = x Wq^T + bq, then row softmax ----
#pragma unroll
  for (int t = 0; t < 8; ++t) acc[t] = zero8();
  for (int kc = 0; kc < D; kc += 32) {
    v16bf a = load_a_g(xb, rowBase, kc, lane, maxRow);
#pragma unroll
    for (int t = 0; t < 8; ++t) acc[t] = WMMA_BF16(a, load_b_l(Wql, t * 16, kc, lane), acc[t]);
  }
#pragma unroll
  for (int t = 0; t < 8; ++t) {
    float bias = bq[t * 16 + l];
#pragma unroll
    for (int g = 0; g < 8; ++g) acc[t][g] += bias;
  }
#pragma unroll
  for (int g = 0; g < 8; ++g) {               // row max across 128 cols
    float m = acc[0][g];
#pragma unroll
    for (int t = 1; t < 8; ++t) m = fmaxf(m, acc[t][g]);
    m = fmaxf(m, __shfl_xor(m, 1));  m = fmaxf(m, __shfl_xor(m, 2));
    m = fmaxf(m, __shfl_xor(m, 4));  m = fmaxf(m, __shfl_xor(m, 8));
    rstat[g] = m;
  }
#pragma unroll
  for (int t = 0; t < 8; ++t)
#pragma unroll
    for (int g = 0; g < 8; ++g) acc[t][g] = __expf(acc[t][g] - rstat[g]);
#pragma unroll
  for (int g = 0; g < 8; ++g) {               // row sum
    float s = 0.f;
#pragma unroll
    for (int t = 0; t < 8; ++t) s += acc[t][g];
    s += __shfl_xor(s, 1); s += __shfl_xor(s, 2);
    s += __shfl_xor(s, 4); s += __shfl_xor(s, 8);
    rstat[g] = 1.0f / s;
  }
#pragma unroll
  for (int t = 0; t < 8; ++t)
#pragma unroll
    for (int g = 0; g < 8; ++g) {
      int row = rowBase + g + hi * 8;
      if (row < N_NODES)
        qout[(size_t)row * D + t * 16 + l] = f2bf(acc[t][g] * rstat[g]);
    }

  // ---- Klog = x Wk^T + bk, per-block column max ----
#pragma unroll
  for (int t = 0; t < 8; ++t) acc[t] = zero8();
  for (int kc = 0; kc < D; kc += 32) {
    v16bf a = load_a_g(xb, rowBase, kc, lane, maxRow);
#pragma unroll
    for (int t = 0; t < 8; ++t) acc[t] = WMMA_BF16(a, load_b_l(Wkl, t * 16, kc, lane), acc[t]);
  }
#pragma unroll
  for (int t = 0; t < 8; ++t) {
    float bias = bk[t * 16 + l];
    float m = acc[t][0] + bias;
#pragma unroll
    for (int g = 1; g < 8; ++g) m = fmaxf(m, acc[t][g] + bias);
    m = fmaxf(m, __shfl_xor(m, 16));          // combine row halves (same column)
    if (hi == 0) red[wave * D + t * 16 + l] = m;
  }
  __syncthreads();
  if (tid < D) {
    float m = red[tid];
#pragma unroll
    for (int w = 1; w < 8; ++w) m = fmaxf(m, red[w * D + tid]);
    partmax[(size_t)blockIdx.x * D + tid] = m;
  }
}

__global__ void k_colmax(const float* __restrict__ partmax, float* __restrict__ colmax) {
  int c = threadIdx.x;                        // 128 threads
  float m = -3.0e38f;
  for (int b = 0; b < NBLK; ++b) m = fmaxf(m, partmax[(size_t)b * D + c]);
  colmax[c] = m;
}

// ---------------------------------------------------------------------------
// Pass 3: recompute Klog, e = exp(Klog-colmax); accumulate colsum, E^T x, E^T q
__global__ __launch_bounds__(256) void k_kvkq(const unsigned short* __restrict__ xb,
                                              const unsigned short* __restrict__ qb,
                                              const float* __restrict__ Wk,
                                              const float* __restrict__ bk,
                                              const float* __restrict__ colmax,
                                              float* __restrict__ colsum,
                                              float* __restrict__ kvacc,
                                              float* __restrict__ kqacc) {
  __shared__ __align__(16) unsigned short Bstage[D * D];  // Wk -> X^T -> Q^T (reused)
  __shared__ __align__(16) unsigned short ET[D * D];      // [d][n] = e[n][d] (bf16)
  __shared__ float red[8 * D];
  int tid = threadIdx.x, lane = tid & 31, wave = tid >> 5;
  int l = lane & 15, hi = lane >> 4;
  int nBase = blockIdx.x * ROWS_PER_BLK;
  int maxRow = N_NODES - 1;

  for (int i = tid; i < D * D; i += 256) Bstage[i] = f2bf(Wk[i]);
  __syncthreads();

  int rowBase = nBase + wave * 16;
  v8f acc[8];
#pragma unroll
  for (int t = 0; t < 8; ++t) acc[t] = zero8();
  for (int kc = 0; kc < D; kc += 32) {
    v16bf a = load_a_g(xb, rowBase, kc, lane, maxRow);
#pragma unroll
    for (int t = 0; t < 8; ++t) acc[t] = WMMA_BF16(a, load_b_l(Bstage, t * 16, kc, lane), acc[t]);
  }
#pragma unroll
  for (int t = 0; t < 8; ++t) {
    int col = t * 16 + l;
    float bias = bk[col], cmx = colmax[col];
    float cs = 0.f;
#pragma unroll
    for (int g = 0; g < 8; ++g) {
      int row = rowBase + g + hi * 8;
      float e = (row < N_NODES) ? __expf(acc[t][g] + bias - cmx) : 0.f;
      cs += e;
      ET[col * D + wave * 16 + g + hi * 8] = f2bf(e);
    }
    cs += __shfl_xor(cs, 16);
    if (hi == 0) red[wave * D + col] = cs;
  }
  __syncthreads();
  if (tid < D) {
    float s = red[tid];
#pragma unroll
    for (int w = 1; w < 8; ++w) s += red[w * D + tid];
    atomicAdd(&colsum[tid], s);
  }
  __syncthreads();

  // kv partial: E^T @ X
  stage_T(xb, Bstage, nBase, tid);
  __syncthreads();
  int dBase = wave * 16;
#pragma unroll
  for (int t = 0; t < 8; ++t) acc[t] = zero8();
  for (int kc = 0; kc < D; kc += 32) {
    v16bf a = load_a_l(ET, dBase, kc, lane);
#pragma unroll
    for (int t = 0; t < 8; ++t) acc[t] = WMMA_BF16(a, load_b_l(Bstage, t * 16, kc, lane), acc[t]);
  }
#pragma unroll
  for (int t = 0; t < 8; ++t)
#pragma unroll
    for (int g = 0; g < 8; ++g)
      atomicAdd(&kvacc[(dBase + g + hi * 8) * D + t * 16 + l], acc[t][g]);
  __syncthreads();

  // kq partial: E^T @ Q
  stage_T(qb, Bstage, nBase, tid);
  __syncthreads();
#pragma unroll
  for (int t = 0; t < 8; ++t) acc[t] = zero8();
  for (int kc = 0; kc < D; kc += 32) {
    v16bf a = load_a_l(ET, dBase, kc, lane);
#pragma unroll
    for (int t = 0; t < 8; ++t) acc[t] = WMMA_BF16(a, load_b_l(Bstage, t * 16, kc, lane), acc[t]);
  }
#pragma unroll
  for (int t = 0; t < 8; ++t)
#pragma unroll
    for (int g = 0; g < 8; ++g)
      atomicAdd(&kqacc[(dBase + g + hi * 8) * D + t * 16 + l], acc[t][g]);
}

// ---------------------------------------------------------------------------
// Pass 4 (one workgroup): normalize, power chain, M^T (bf16) out
__global__ __launch_bounds__(256) void k_chain(const float* __restrict__ kvacc,
                                               const float* __restrict__ kqacc,
                                               const float* __restrict__ colsum,
                                               const float* __restrict__ temp,
                                               unsigned short* __restrict__ MbT) {
  __shared__ __align__(16) unsigned short curb[D * D];   // cur (bf16, row-major)
  __shared__ __align__(16) unsigned short kqT[D * D];    // kq^T (bf16)
  int tid = threadIdx.x, lane = tid & 31, wave = tid >> 5;
  int l = lane & 15, hi = lane >> 4;
  float macc[64];                                        // element i = tid + 256*j
#pragma unroll
  for (int j = 0; j < 64; ++j) macc[j] = 0.f;

  for (int i = tid; i < D * D; i += 256) {
    int d1 = i >> 7, d2 = i & 127;
    float inv = 1.0f / colsum[d1];
    curb[i] = f2bf(kvacc[i] * inv);                      // kv
    kqT[d2 * D + d1] = f2bf(kqacc[i] * inv);             // kq transposed
  }
  __syncthreads();

  for (int it = 1; it <= KPOW; ++it) {
    float gamma = temp[it];
#pragma unroll
    for (int j = 0; j < 64; ++j) macc[j] += gamma * bf2f(curb[tid + 256 * j]);
    if (it < KPOW) {
      v8f acc[8];
#pragma unroll
      for (int t = 0; t < 8; ++t) acc[t] = zero8();
      int rBase = wave * 16;
      for (int kc = 0; kc < D; kc += 32) {
        v16bf a = load_a_l(curb, rBase, kc, lane);
#pragma unroll
        for (int t = 0; t < 8; ++t) acc[t] = WMMA_BF16(a, load_b_l(kqT, t * 16, kc, lane), acc[t]);
      }
      __syncthreads();                                   // all reads of curb done
#pragma unroll
      for (int t = 0; t < 8; ++t)
#pragma unroll
        for (int g = 0; g < 8; ++g)
          curb[(rBase + g + hi * 8) * D + t * 16 + l] = f2bf(acc[t][g]);
      __syncthreads();
    }
  }
#pragma unroll
  for (int j = 0; j < 64; ++j) {
    int i = tid + 256 * j;
    MbT[(i & 127) * D + (i >> 7)] = f2bf(macc[j]);       // store M transposed
  }
}

// ---------------------------------------------------------------------------
// Pass 5: out = temp0*x + q @ M
__global__ __launch_bounds__(256) void k_out(const unsigned short* __restrict__ qb,
                                             const unsigned short* __restrict__ MbT,
                                             const float* __restrict__ x,
                                             const float* __restrict__ temp,
                                             float* __restrict__ out) {
  __shared__ __align__(16) unsigned short MT[D * D];
  int tid = threadIdx.x, lane = tid & 31, wave = tid >> 5;
  int l = lane & 15, hi = lane >> 4;
  for (int i = tid; i < D * D; i += 256) MT[i] = MbT[i];
  __syncthreads();
  float t0 = temp[0];
  int rowBase = blockIdx.x * ROWS_PER_BLK + wave * 16;
  int maxRow = N_NODES - 1;
  v8f acc[8];
#pragma unroll
  for (int t = 0; t < 8; ++t) acc[t] = zero8();
  for (int kc = 0; kc < D; kc += 32) {
    v16bf a = load_a_g(qb, rowBase, kc, lane, maxRow);
#pragma unroll
    for (int t = 0; t < 8; ++t) acc[t] = WMMA_BF16(a, load_b_l(MT, t * 16, kc, lane), acc[t]);
  }
#pragma unroll
  for (int t = 0; t < 8; ++t)
#pragma unroll
    for (int g = 0; g < 8; ++g) {
      int row = rowBase + g + hi * 8;
      if (row < N_NODES) {
        size_t idx = (size_t)row * D + t * 16 + l;
        out[idx] = acc[t][g] + t0 * x[idx];
      }
    }
}

// ---------------------------------------------------------------------------
extern "C" void kernel_launch(void* const* d_in, const int* in_sizes, int n_in,
                              void* d_out, int out_size, void* d_ws, size_t ws_size,
                              hipStream_t stream) {
  (void)in_sizes; (void)n_in; (void)out_size; (void)ws_size;
  const float* x    = (const float*)d_in[0];
  const float* Wq   = (const float*)d_in[1];
  const float* bq   = (const float*)d_in[2];
  const float* Wk   = (const float*)d_in[3];
  const float* bk   = (const float*)d_in[4];
  const float* temp = (const float*)d_in[5];
  float* out = (float*)d_out;

  char* ws = (char*)d_ws;
  size_t off = 0;
  auto alloc = [&](size_t bytes) -> void* {
    off = (off + 255) & ~(size_t)255;
    void* p = ws + off;
    off += bytes;
    return p;
  };
  unsigned short* xb      = (unsigned short*)alloc((size_t)N_NODES * D * 2);
  unsigned short* qb      = (unsigned short*)alloc((size_t)N_NODES * D * 2);
  float*          partmax = (float*)alloc((size_t)NBLK * D * 4);
  float*          colmax  = (float*)alloc(D * 4);
  float*          zreg    = (float*)alloc((size_t)(D + 2 * D * D) * 4);
  float*          colsum  = zreg;
  float*          kvacc   = zreg + D;
  float*          kqacc   = kvacc + D * D;
  unsigned short* MbT     = (unsigned short*)alloc(D * D * 2);

  int nconv = (N_NODES * D) / 8;                    // 3,200,000 threads, exact
  k_convert<<<(nconv + 255) / 256, 256, 0, stream>>>(x, xb);
  int nz = D + 2 * D * D;
  k_zero<<<(nz + 255) / 256, 256, 0, stream>>>(zreg, nz);
  k_qk<<<NBLK, 256, 0, stream>>>(xb, Wq, bq, Wk, bk, qb, partmax);
  k_colmax<<<1, 128, 0, stream>>>(partmax, colmax);
  k_kvkq<<<NBLK, 256, 0, stream>>>(xb, qb, Wk, bk, colmax, colsum, kvacc, kqacc);
  k_chain<<<1, 256, 0, stream>>>(kvacc, kqacc, colsum, temp, MbT);
  k_out<<<NBLK, 256, 0, stream>>>(qb, MbT, x, temp, out);
}
